// AttentionLayer_3985729650926
// MI455X (gfx1250) — matrix operand order repaired and checked by reading, not compiled, not run on hardware
//
#include <hip/hip_runtime.h>
#include <hip/hip_bf16.h>
#include <math.h>

// Problem constants
#define LL   512
#define BB   8
#define EE   1024
#define HH   16
#define DD   64
#define HIDN 4096
#define NEXP 5
#define CAPS 4
#define NTOK (LL*BB)   // 4096

typedef __bf16 bf16_t;
typedef __attribute__((ext_vector_type(16))) __bf16 v16bf;
typedef __attribute__((ext_vector_type(8)))  __bf16 v8bf;
typedef __attribute__((ext_vector_type(8)))  float  v8f;

static __device__ __forceinline__ v16bf cat8(v8bf lo, v8bf hi) {
  return __builtin_shufflevector(lo, hi, 0,1,2,3,4,5,6,7,8,9,10,11,12,13,14,15);
}

// ---------------------------------------------------------------- convert
__global__ __launch_bounds__(256) void k_f32_to_bf16(const float* __restrict__ in,
                                                     bf16_t* __restrict__ out, long n) {
  long i = (long)blockIdx.x * blockDim.x + threadIdx.x;
  long stride = (long)gridDim.x * blockDim.x;
  for (; i < n; i += stride) out[i] = (bf16_t)in[i];
}

// ---------------------------------------------------------------- WMMA GEMM
// C(MxN) = A(MxK) * B  (+bias, +gelu, +residual).  A: bf16 row-major.
// BT=true : B given as (N x K) row-major (i.e. X @ W^T with W=(N,K))
// BT=false: B given as (K x N) row-major
// 64x64 tile per 128-thread block (4 waves); wave w owns N-subtile w; BK=32.
template<bool BT, bool GELU, bool BIAS, bool RES, bool OF32, bool OBF>
__global__ __launch_bounds__(128) void k_gemm(
    const bf16_t* __restrict__ A, const bf16_t* __restrict__ Bm,
    const float* __restrict__ bias, const float* __restrict__ res,
    float* __restrict__ outF, bf16_t* __restrict__ outB,
    int M, int N, int K)
{
  __shared__ bf16_t As[64*32];   // [m][k]
  __shared__ bf16_t Bs[64*32];   // [n][k]
  const int n0   = blockIdx.x * 64;
  const int m0   = blockIdx.y * 64;
  const int tid  = threadIdx.x;
  const int wave = tid >> 5;
  const int lane = tid & 31;
  const int hf   = lane >> 4;    // lane half: 0 or 1
  const int l15  = lane & 15;

  v8f acc[4] = {};               // 4 M-subtiles x (16x16 f32)

  for (int k0 = 0; k0 < K; k0 += 32) {
    { // A tile: 64x32, each thread copies 8+8 bf16
      int r = tid >> 1, c0 = (tid & 1) * 16;
      const bf16_t* src = A + (size_t)(m0 + r) * K + k0 + c0;
      *(v8bf*)&As[r*32 + c0]     = *(const v8bf*)src;
      *(v8bf*)&As[r*32 + c0 + 8] = *(const v8bf*)(src + 8);
      if (k0 + 32 < K) __builtin_prefetch(src + 32, 0, 1);   // global_prefetch_b8
    }
    if (BT) { // B is (N x K): straight copy into Bs[n][k]
      int r = tid >> 1, c0 = (tid & 1) * 16;
      const bf16_t* src = Bm + (size_t)(n0 + r) * K + k0 + c0;
      *(v8bf*)&Bs[r*32 + c0]     = *(const v8bf*)src;
      *(v8bf*)&Bs[r*32 + c0 + 8] = *(const v8bf*)(src + 8);
      if (k0 + 32 < K) __builtin_prefetch(src + 32, 0, 1);
    } else {  // B is (K x N): transpose into Bs[n][k]
      int kk = tid >> 2, c0 = (tid & 3) * 16;
      const bf16_t* src = Bm + (size_t)(k0 + kk) * N + n0 + c0;
      v8bf b0 = *(const v8bf*)src;
      v8bf b1 = *(const v8bf*)(src + 8);
#pragma unroll
      for (int j = 0; j < 8; ++j) Bs[(c0 + j)*32 + kk]     = b0[j];
#pragma unroll
      for (int j = 0; j < 8; ++j) Bs[(c0 + 8 + j)*32 + kk] = b1[j];
      if (k0 + 32 < K) __builtin_prefetch(src + (size_t)32 * N, 0, 1);
    }
    __syncthreads();

    // B fragment (32x16): lane half 0 -> K 0..15, half 1 -> K 16..31, col = l15
    const bf16_t* bp = &Bs[(wave*16 + l15)*32 + hf*16];
    v16bf bfrag = cat8(*(const v8bf*)bp, *(const v8bf*)(bp + 8));
#pragma unroll
    for (int ms = 0; ms < 4; ++ms) {
      // A fragment (16x32): row = l15; half0: K 0..15; half1: K 16..31
      const bf16_t* ap = &As[(ms*16 + l15)*32 + hf*16];
      v16bf afrag = cat8(*(const v8bf*)ap, *(const v8bf*)(ap + 8));
      acc[ms] = __builtin_amdgcn_wmma_f32_16x16x32_bf16(
          false, afrag, false, bfrag, (short)0, acc[ms], false, false);
    }
    __syncthreads();
  }

  const int col = n0 + wave*16 + l15;
  float bv = 0.0f;
  if (BIAS) bv = bias[col];
#pragma unroll
  for (int ms = 0; ms < 4; ++ms) {
#pragma unroll
    for (int i = 0; i < 8; ++i) {
      int row = m0 + ms*16 + hf*8 + i;     // C/D layout: M = i + 8*half
      float v = acc[ms][i] + bv;
      if (GELU) v = 0.5f * v * (1.0f + erff(v * 0.70710678118f));
      if (RES)  v += res[(size_t)row * N + col];
      if (OF32) outF[(size_t)row * N + col] = v;
      if (OBF)  outB[(size_t)row * N + col] = (bf16_t)v;
    }
  }
}

// ---------------------------------------------------------------- flash attention (WMMA, LDS-staged)
// grid: (B*H, L/64), block 128 (4 waves). Wave w: 16 queries qw..qw+15.
// qkv layout: (L*B, 3E) bf16; q at +0, k at +E, v at +2E, head h at +h*64.
// K/V tiles (32 keys) are cooperatively staged in LDS with 16B coalesced loads;
// V is transposed during staging so every fragment read is a pair of ds_load_b128.
__global__ __launch_bounds__(128) void k_attn(
    const bf16_t* __restrict__ qkv, const float* __restrict__ tm,
    bf16_t* __restrict__ attnb)
{
  __shared__ bf16_t Ks[32*64];      // [key][d]
  __shared__ bf16_t Vt[64*32];      // [d][key]   (transposed)
  __shared__ bf16_t Pst[4][16*32];  // wave-private P staging
  const int bh   = blockIdx.x;
  const int b    = bh >> 4;          // H = 16
  const int h    = bh & 15;
  const int wave = threadIdx.x >> 5;
  const int lane = threadIdx.x & 31;
  const int hf   = lane >> 4;
  const int l15  = lane & 15;
  const int q0   = blockIdx.y * 64;
  const int qw   = q0 + wave * 16;

  // Q fragments for K-steps d=[0,32) and [32,64)
  const bf16_t* qrow = qkv + ((size_t)(qw + l15) * BB + b) * (3*EE) + h * DD;
  v16bf Aq0, Aq1;
  { const bf16_t* p = qrow + 0*32 + hf*16; Aq0 = cat8(*(const v8bf*)p, *(const v8bf*)(p+8)); }
  { const bf16_t* p = qrow + 1*32 + hf*16; Aq1 = cat8(*(const v8bf*)p, *(const v8bf*)(p+8)); }

  float tq[8];
#pragma unroll
  for (int i = 0; i < 8; ++i) tq[i] = tm[(size_t)(qw + hf*8 + i) * BB + b];

  float mrun[8], lrun[8];
#pragma unroll
  for (int i = 0; i < 8; ++i) { mrun[i] = -3.0e38f; lrun[i] = 0.0f; }
  v8f O0 = {}, O1 = {}, O2 = {}, O3 = {};

  const int ktend = (q0 + 63) >> 5;   // uniform across block (barrier-safe)
  const int ktmax = (qw + 15) >> 5;   // this wave's causal frontier

  for (int kt = 0; kt <= ktend; ++kt) {
    // ---- cooperative staging: 32-key K tile and transposed V tile ----
    {
      const int t = threadIdx.x;
#pragma unroll
      for (int c = 0; c < 2; ++c) {       // 256 chunks of 8 bf16, 2 per thread
        int ch  = t + c * 128;
        int key = ch >> 3;                // 8 chunks per 64-wide row
        int d0  = (ch & 7) * 8;
        size_t rowb = ((size_t)(kt*32 + key) * BB + b) * (3*EE) + h*DD + d0;
        *(v8bf*)&Ks[key*64 + d0] = *(const v8bf*)(qkv + rowb + EE);
        v8bf vv = *(const v8bf*)(qkv + rowb + 2*EE);
#pragma unroll
        for (int j = 0; j < 8; ++j) Vt[(d0 + j)*32 + key] = vv[j];
      }
    }
    __syncthreads();

    if (kt <= ktmax) {
      // ---- scores: S(16q x 32keys) via 4 WMMAs from LDS ----
      v8f S0 = {}, S1 = {};
#pragma unroll
      for (int ks = 0; ks < 2; ++ks) {
        v16bf aq = ks ? Aq1 : Aq0;
        const bf16_t* kp0 = &Ks[l15*64        + ks*32 + hf*16];
        v16bf bk0 = cat8(*(const v8bf*)kp0, *(const v8bf*)(kp0+8));
        S0 = __builtin_amdgcn_wmma_f32_16x16x32_bf16(false, aq, false, bk0, (short)0, S0, false, false);
        const bf16_t* kp1 = &Ks[(16+l15)*64   + ks*32 + hf*16];
        v16bf bk1 = cat8(*(const v8bf*)kp1, *(const v8bf*)(kp1+8));
        S1 = __builtin_amdgcn_wmma_f32_16x16x32_bf16(false, aq, false, bk1, (short)0, S1, false, false);
      }
      const float tk0 = tm[(size_t)(kt*32 + l15) * BB + b];
      const float tk1 = tm[(size_t)(kt*32 + 16 + l15) * BB + b];
      float p0[8], p1[8], mt[8], st[8], scale[8];
#pragma unroll
      for (int i = 0; i < 8; ++i) {
        float s0 = (tk0 <= tq[i]) ? S0[i] * 0.125f : -1.0e30f;  // 1/sqrt(64)
        float s1 = (tk1 <= tq[i]) ? S1[i] * 0.125f : -1.0e30f;
        p0[i] = s0; p1[i] = s1;
        mt[i] = fmaxf(s0, s1);
      }
#pragma unroll
      for (int i = 0; i < 8; ++i) {       // row-max across 16 lanes of same half
        mt[i] = fmaxf(mt[i], __shfl_xor(mt[i], 1, 32));
        mt[i] = fmaxf(mt[i], __shfl_xor(mt[i], 2, 32));
        mt[i] = fmaxf(mt[i], __shfl_xor(mt[i], 4, 32));
        mt[i] = fmaxf(mt[i], __shfl_xor(mt[i], 8, 32));
      }
#pragma unroll
      for (int i = 0; i < 8; ++i) {
        float nm = fmaxf(mrun[i], mt[i]);
        scale[i] = __expf(mrun[i] - nm);
        p0[i] = __expf(p0[i] - nm);
        p1[i] = __expf(p1[i] - nm);
        st[i] = p0[i] + p1[i];
        mrun[i] = nm;
      }
#pragma unroll
      for (int i = 0; i < 8; ++i) {
        st[i] += __shfl_xor(st[i], 1, 32);
        st[i] += __shfl_xor(st[i], 2, 32);
        st[i] += __shfl_xor(st[i], 4, 32);
        st[i] += __shfl_xor(st[i], 8, 32);
        lrun[i] = lrun[i] * scale[i] + st[i];
      }
#pragma unroll
      for (int i = 0; i < 8; ++i) {
        O0[i] *= scale[i]; O1[i] *= scale[i]; O2[i] *= scale[i]; O3[i] *= scale[i];
      }
      // P (C-layout) -> wave-private LDS -> A-layout fragment (same-wave DS in-order)
      bf16_t* pw = &Pst[wave][0];
#pragma unroll
      for (int i = 0; i < 8; ++i) {
        pw[(hf*8 + i)*32 + l15]      = (bf16_t)p0[i];
        pw[(hf*8 + i)*32 + 16 + l15] = (bf16_t)p1[i];
      }
      const bf16_t* pp = &pw[l15*32 + hf*16];
      v16bf Ap = cat8(*(const v8bf*)pp, *(const v8bf*)(pp+8));
      // V fragments from transposed LDS tile: B[k=key][n=d]
#pragma unroll
      for (int ds = 0; ds < 4; ++ds) {
        const bf16_t* vp = &Vt[(ds*16 + l15)*32 + hf*16];
        v16bf Bv = cat8(*(const v8bf*)vp, *(const v8bf*)(vp+8));
        if (ds == 0) O0 = __builtin_amdgcn_wmma_f32_16x16x32_bf16(false, Ap, false, Bv, (short)0, O0, false, false);
        if (ds == 1) O1 = __builtin_amdgcn_wmma_f32_16x16x32_bf16(false, Ap, false, Bv, (short)0, O1, false, false);
        if (ds == 2) O2 = __builtin_amdgcn_wmma_f32_16x16x32_bf16(false, Ap, false, Bv, (short)0, O2, false, false);
        if (ds == 3) O3 = __builtin_amdgcn_wmma_f32_16x16x32_bf16(false, Ap, false, Bv, (short)0, O3, false, false);
      }
    }
    __syncthreads();
  }

#pragma unroll
  for (int i = 0; i < 8; ++i) {
    float inv = 1.0f / lrun[i];
    size_t base = ((size_t)(qw + hf*8 + i) * BB + b) * EE + h * DD;
    attnb[base +  0 + l15] = (bf16_t)(O0[i] * inv);
    attnb[base + 16 + l15] = (bf16_t)(O1[i] * inv);
    attnb[base + 32 + l15] = (bf16_t)(O2[i] * inv);
    attnb[base + 48 + l15] = (bf16_t)(O3[i] * inv);
  }
}

// ---------------------------------------------------------------- layernorm
__global__ __launch_bounds__(256) void k_ln(const float* __restrict__ in,
    const float* __restrict__ gw, const float* __restrict__ bw,
    float* __restrict__ out, bf16_t* __restrict__ outb)
{
  __shared__ float sh[16];
  const int row = blockIdx.x, tid = threadIdx.x;
  const float* p = in + (size_t)row * EE;
  float v[4], s = 0.0f, sq = 0.0f;
#pragma unroll
  for (int i = 0; i < 4; ++i) { v[i] = p[tid + i*256]; s += v[i]; sq += v[i]*v[i]; }
#pragma unroll
  for (int off = 16; off >= 1; off >>= 1) {
    s  += __shfl_xor(s,  off, 32);
    sq += __shfl_xor(sq, off, 32);
  }
  if ((tid & 31) == 0) { sh[tid>>5] = s; sh[8 + (tid>>5)] = sq; }
  __syncthreads();
  float ts = 0.0f, tsq = 0.0f;
#pragma unroll
  for (int w = 0; w < 8; ++w) { ts += sh[w]; tsq += sh[8+w]; }
  float mu  = ts * (1.0f/EE);
  float var = tsq * (1.0f/EE) - mu*mu;
  float rs  = rsqrtf(var + 1e-5f);
#pragma unroll
  for (int i = 0; i < 4; ++i) {
    int c = tid + i*256;
    float o = (v[i]-mu)*rs*gw[c] + bw[c];
    out[(size_t)row*EE + c] = o;
    if (outb) outb[(size_t)row*EE + c] = (bf16_t)o;
  }
}

// ---------------------------------------------------------------- gating softmax (wave/token)
__global__ __launch_bounds__(128) void k_gate(const float* __restrict__ x1,
    const float* __restrict__ gw, float* __restrict__ raw)
{
  int tok  = blockIdx.x * 4 + (threadIdx.x >> 5);
  int lane = threadIdx.x & 31;
  float s[NEXP] = {};
  const float* xr = x1 + (size_t)tok * EE;
  for (int e = lane; e < EE; e += 32) {
    float xv = xr[e];
#pragma unroll
    for (int j = 0; j < NEXP; ++j) s[j] += xv * gw[e*NEXP + j];
  }
#pragma unroll
  for (int j = 0; j < NEXP; ++j) {
#pragma unroll
    for (int off = 16; off >= 1; off >>= 1) s[j] += __shfl_xor(s[j], off, 32);
  }
  if (lane == 0) {
    float mx = s[0];
#pragma unroll
    for (int j = 1; j < NEXP; ++j) mx = fmaxf(mx, s[j]);
    float ex[NEXP], den = 0.0f;
#pragma unroll
    for (int j = 0; j < NEXP; ++j) { ex[j] = __expf(s[j]-mx); den += ex[j]; }
#pragma unroll
    for (int j = 0; j < NEXP; ++j) raw[(size_t)tok*NEXP + j] = ex[j]/den;
  }
}

// ---------------------------------------------------------------- top-2 gating + capacity (per group)
__global__ __launch_bounds__(256) void k_top2(const float* __restrict__ raw,
    int* __restrict__ slot_src, int* __restrict__ me, int* __restrict__ mc,
    float* __restrict__ mg)
{
  int g = blockIdx.x * blockDim.x + threadIdx.x;
  if (g >= LL) return;
  for (int e = 0; e < NEXP; ++e)
    for (int c = 0; c < CAPS; ++c)
      slot_src[((size_t)e*LL + g)*CAPS + c] = -1;

  int   c1raw[NEXP] = {0,0,0,0,0};
  int   e1a[BB], e2a[BB], p1a[BB];
  float g1a[BB], g2a[BB];
  for (int n = 0; n < BB; ++n) {
    float r[NEXP];
    for (int j = 0; j < NEXP; ++j) r[j] = raw[((size_t)g*BB + n)*NEXP + j];
    int e1 = 0;
    for (int j = 1; j < NEXP; ++j) if (r[j] > r[e1]) e1 = j;   // first argmax
    float g1 = r[e1];
    int e2 = (e1 == 0) ? 1 : 0; float g2 = r[e2];
    for (int j = 0; j < NEXP; ++j) { if (j == e1 || j == e2) continue; if (r[j] > g2) { g2 = r[j]; e2 = j; } }
    float den = g1 + g2 + 1e-9f;
    g1a[n] = g1/den; g2a[n] = g2/den; e1a[n] = e1; e2a[n] = e2;
    p1a[n] = c1raw[e1];           // exclusive cumsum of (unmasked) m1
    c1raw[e1]++;
  }
  int cnt1[NEXP];
  for (int j = 0; j < NEXP; ++j) cnt1[j] = (c1raw[j] < CAPS) ? c1raw[j] : CAPS;
  int c2raw[NEXP] = {0,0,0,0,0};
  for (int n = 0; n < BB; ++n) {
    int tok = g*BB + n;
    int e1 = e1a[n], p1 = p1a[n];
    int keep1 = (p1 < CAPS);
    me[tok] = e1; mc[tok] = keep1 ? p1 : 0; mg[tok] = keep1 ? g1a[n] : 0.0f;
    if (keep1) slot_src[((size_t)e1*LL + g)*CAPS + p1] = n;
    int e2 = e2a[n];
    int p2 = c2raw[e2] + cnt1[e2];
    c2raw[e2]++;
    int keep2 = (p2 < CAPS);
    me[NTOK + tok] = e2; mc[NTOK + tok] = keep2 ? p2 : 0; mg[NTOK + tok] = keep2 ? g2a[n] : 0.0f;
    if (keep2) slot_src[((size_t)e2*LL + g)*CAPS + p2] = n;
  }
}

// ---------------------------------------------------------------- dispatch gather -> bf16
__global__ __launch_bounds__(256) void k_dispatch(const int* __restrict__ slot_src,
    const float* __restrict__ x1, bf16_t* __restrict__ einb)
{
  int slot = blockIdx.x;                 // (e*LL + g)*CAPS + c
  int g    = (slot >> 2) % LL;
  int src  = slot_src[slot];
  bf16_t* dst = einb + (size_t)slot * EE;
  if (src < 0) {
    for (int i = threadIdx.x; i < EE; i += 256) dst[i] = (bf16_t)0.0f;
  } else {
    const float* s = x1 + ((size_t)g * BB + src) * EE;
    for (int i = threadIdx.x; i < EE; i += 256) dst[i] = (bf16_t)s[i];
  }
}

// ---------------------------------------------------------------- combine + residual
__global__ __launch_bounds__(256) void k_combine(const float* __restrict__ eo,
    const int* __restrict__ me, const int* __restrict__ mc, const float* __restrict__ mg,
    const float* __restrict__ x1, float* __restrict__ y2)
{
  int tok = blockIdx.x;
  int g   = tok >> 3;                    // B = 8
  int   e1 = me[tok],        c1 = mc[tok];        float w1 = mg[tok];
  int   e2 = me[NTOK + tok], c2 = mc[NTOK + tok]; float w2 = mg[NTOK + tok];
  const float* r1 = eo + (((size_t)e1 * LL + g) * CAPS + c1) * EE;
  const float* r2 = eo + (((size_t)e2 * LL + g) * CAPS + c2) * EE;
  const float* xr = x1 + (size_t)tok * EE;
  float*       yr = y2 + (size_t)tok * EE;
  for (int i = threadIdx.x; i < EE; i += 256)
    yr[i] = w1 * r1[i] + w2 * r2[i] + xr[i];
}

// ---------------------------------------------------------------- launcher
extern "C" void kernel_launch(void* const* d_in, const int* in_sizes, int n_in,
                              void* d_out, int out_size, void* d_ws, size_t ws_size,
                              hipStream_t stream)
{
  const float* x      = (const float*)d_in[0];
  const float* tm     = (const float*)d_in[1];
  const float* w_qkv  = (const float*)d_in[2];
  const float* b_qkv  = (const float*)d_in[3];
  const float* w_out  = (const float*)d_in[4];
  const float* b_out  = (const float*)d_in[5];
  const float* ln1g   = (const float*)d_in[6];
  const float* ln1b   = (const float*)d_in[7];
  const float* ln2g   = (const float*)d_in[8];
  const float* ln2b   = (const float*)d_in[9];
  const float* gate_w = (const float*)d_in[10];
  const float* w1     = (const float*)d_in[11];
  const float* w2     = (const float*)d_in[12];
  float* out = (float*)d_out;
  (void)in_sizes; (void)n_in; (void)out_size; (void)ws_size;

  char* ws = (char*)d_ws;
  size_t off = 0;
  auto alloc = [&](size_t bytes) -> void* {
    void* p = ws + off; off = (off + bytes + 255) & ~(size_t)255; return p;
  };

  bf16_t* xb    = (bf16_t*)alloc((size_t)NTOK*EE*2);
  bf16_t* wqkvb = (bf16_t*)alloc((size_t)3*EE*EE*2);
  bf16_t* woutb = (bf16_t*)alloc((size_t)EE*EE*2);
  bf16_t* w1b   = (bf16_t*)alloc((size_t)NEXP*EE*HIDN*2);
  bf16_t* w2b   = (bf16_t*)alloc((size_t)NEXP*HIDN*EE*2);
  bf16_t* qkvb  = (bf16_t*)alloc((size_t)NTOK*3*EE*2);
  bf16_t* attnb = (bf16_t*)alloc((size_t)NTOK*EE*2);
  float*  y     = (float*)alloc((size_t)NTOK*EE*4);
  float*  x1    = (float*)alloc((size_t)NTOK*EE*4);
  float*  raw   = (float*)alloc((size_t)NTOK*NEXP*4);
  int*    slot  = (int*)alloc((size_t)NEXP*LL*CAPS*4);
  int*    me    = (int*)alloc((size_t)2*NTOK*4);
  int*    mc    = (int*)alloc((size_t)2*NTOK*4);
  float*  mg    = (float*)alloc((size_t)2*NTOK*4);
  bf16_t* einb  = (bf16_t*)alloc((size_t)NEXP*LL*CAPS*EE*2);
  bf16_t* hb    = (bf16_t*)alloc((size_t)LL*CAPS*HIDN*2);       // reused per expert
  float*  eo    = (float*)alloc((size_t)NEXP*LL*CAPS*EE*4);
  float*  y2    = (float*)alloc((size_t)NTOK*EE*4);

  // 1. bf16 conversions
  k_f32_to_bf16<<<1024, 256, 0, stream>>>(x,     xb,    (long)NTOK*EE);
  k_f32_to_bf16<<<1024, 256, 0, stream>>>(w_qkv, wqkvb, (long)3*EE*EE);
  k_f32_to_bf16<<<1024, 256, 0, stream>>>(w_out, woutb, (long)EE*EE);
  k_f32_to_bf16<<<2048, 256, 0, stream>>>(w1,    w1b,   (long)NEXP*EE*HIDN);
  k_f32_to_bf16<<<2048, 256, 0, stream>>>(w2,    w2b,   (long)NEXP*HIDN*EE);

  // 2. QKV projection: (4096x1024) @ (3072x1024)^T + b_qkv -> qkvb (bf16)
  k_gemm<true,false,true,false,false,true><<<dim3(3*EE/64, NTOK/64), dim3(128), 0, stream>>>(
      xb, wqkvb, b_qkv, nullptr, nullptr, qkvb, NTOK, 3*EE, EE);

  // 3. Temporal flash attention (WMMA) -> attnb (L,B,E) bf16
  k_attn<<<dim3(BB*HH, LL/64), dim3(128), 0, stream>>>(qkvb, tm, attnb);

  // 4. Out projection + bias + residual(x) -> y (f32)
  k_gemm<true,false,true,true,true,false><<<dim3(EE/64, NTOK/64), dim3(128), 0, stream>>>(
      attnb, woutb, b_out, x, y, nullptr, NTOK, EE, EE);

  // 5. LayerNorm 1 -> x1 (f32)
  k_ln<<<NTOK, 256, 0, stream>>>(y, ln1g, ln1b, x1, nullptr);

  // 6. MoE gating
  k_gate<<<NTOK/4, 128, 0, stream>>>(x1, gate_w, raw);
  k_top2<<<2, 256, 0, stream>>>(raw, slot, me, mc, mg);
  k_dispatch<<<NEXP*LL*CAPS, 256, 0, stream>>>(slot, x1, einb);

  // 7. Expert FFNs (per expert; h buffer reused)
  for (int e = 0; e < NEXP; ++e) {
    k_gemm<false,true,false,false,false,true><<<dim3(HIDN/64, (LL*CAPS)/64), dim3(128), 0, stream>>>(
        einb + (size_t)e*LL*CAPS*EE, w1b + (size_t)e*EE*HIDN,
        nullptr, nullptr, nullptr, hb, LL*CAPS, HIDN, EE);
    k_gemm<false,false,false,false,true,false><<<dim3(EE/64, (LL*CAPS)/64), dim3(128), 0, stream>>>(
        hb, w2b + (size_t)e*HIDN*EE,
        nullptr, nullptr, eo + (size_t)e*LL*CAPS*EE, nullptr, LL*CAPS, EE, HIDN);
  }

  // 8. Combine + residual(x1) -> y2, then LayerNorm 2 -> out
  k_combine<<<NTOK, 256, 0, stream>>>(eo, me, mc, mg, x1, y2);
  k_ln<<<NTOK, 256, 0, stream>>>(y2, ln2g, ln2b, out, nullptr);
}
